// Net_56118042689827
// MI455X (gfx1250) — matrix operand order repaired and checked by reading, not compiled, not run on hardware
//
#include <hip/hip_runtime.h>

#define NN 50000
#define EE 800000
#define GG 256
#define LL 4
#define TT 4
#define FF 32
#define EPSF 1e-5f

typedef __attribute__((ext_vector_type(16))) _Float16 v16h;
typedef __attribute__((ext_vector_type(8)))  float    v8f;
typedef __attribute__((ext_vector_type(4)))  unsigned v4u;
typedef __attribute__((ext_vector_type(8)))  int      v8i;
typedef __attribute__((ext_vector_type(4)))  int      v4i;

__device__ __forceinline__ v8f wmma16(v16h a, v16h b, v8f c) {
  return __builtin_amdgcn_wmma_f32_16x16x32_f16(false, a, false, b, (short)0, c,
                                                false, false);
}

// monotone float <-> ordered-int bijection (self-inverse) for atomic min/max
__device__ __forceinline__ int f2ord(float f) {
  int i = __float_as_int(f);
  return i >= 0 ? i : (i ^ 0x7fffffff);
}
__device__ __forceinline__ float ord2f(int i) {
  return __int_as_float(i >= 0 ? i : (i ^ 0x7fffffff));
}

// A-matrix (16x32 f16) loader per CDNA5 ISA layout, from f32 row (ld=32).
__device__ __forceinline__ v16h load_a_row(const float* rowptr, int kb) {
  v16h a;
#pragma unroll
  for (int i = 0; i < 8; ++i) {
    a[i]     = (_Float16)rowptr[kb + i];
    a[8 + i] = (_Float16)rowptr[kb + 16 + i];
  }
  return a;
}

// ---------------------------------------------------------------- setup
__global__ void k_zero_deg(int* deg) {
  int i = blockIdx.x * blockDim.x + threadIdx.x;
  if (i < NN) deg[i] = 0;
}

__global__ void k_deg(const int* dst, int* deg) {
  int e = blockIdx.x * blockDim.x + threadIdx.x;
  if (e < EE) atomicAdd(&deg[dst[e]], 1);
}

__global__ void k_nodeprep(const int* deg, const float* avg_log, float* dinv,
                           float* amp, float* att, float* hasf) {
  int n = blockIdx.x * blockDim.x + threadIdx.x;
  if (n >= NN) return;
  float d  = (float)deg[n];
  float dc = fmaxf(d, 1.f);
  float ld = logf(dc + 1.f);
  float av = avg_log[0];
  dinv[n] = 1.f / dc;
  amp[n]  = ld / av;
  att[n]  = av / ld;
  hasf[n] = d > 0.f ? 1.f : 0.f;
}

__global__ void k_embed(const int* x, const float* emb, float* h) {
  int i = blockIdx.x * blockDim.x + threadIdx.x;
  if (i >= NN * FF) return;
  int n = i >> 5, c = i & 31;
  h[i] = emb[x[n] * FF + c];
}

// ------------------------------------------- per-layer edge-type table
// cL[a][t*F+o] = (edge_emb[a] @ enc_w[l] + enc_b[l]) @ W3[t] + pre_b[l][t]
__global__ void k_ctab(const float* edge_emb, const float* enc_w,
                       const float* enc_b, const float* pre_w,
                       const float* pre_b, float* cL, int l) {
  __shared__ float se[FF];
  int a = blockIdx.x, tid = threadIdx.x;
  if (tid < FF) {
    float s = enc_b[l * FF + tid];
    for (int k = 0; k < 50; ++k)
      s += edge_emb[a * 50 + k] * enc_w[(l * 50 + k) * FF + tid];
    se[tid] = s;
  }
  __syncthreads();
  int t = tid >> 5, o = tid & 31;
  float s = pre_b[((l * TT) + t) * FF + o];
  for (int j = 0; j < FF; ++j)
    s += se[j] * pre_w[(((size_t)(l * TT + t)) * 96 + 64 + j) * FF + o];
  cL[a * 128 + tid] = s;
}

// ---------------------- P[N,256] = h[N,32] @ [W1cat | W2cat]  (WMMA)
__global__ void k_gemm_pre(const float* h, const float* pre_w, float* P, int l) {
  int wid  = (blockIdx.x * blockDim.x + threadIdx.x) >> 5;
  int lane = threadIdx.x & 31;
  int mt = wid >> 4, nt = wid & 15;
  int m0 = mt * 16;
  int r = lane & 15;
  int kbA = (lane < 16) ? 0 : 8;
  int kbB = (lane < 16) ? 0 : 8;
  v16h A = load_a_row(h + (size_t)(m0 + r) * FF, kbA);
  int n = nt * 16 + r;                // global column 0..255
  int half = n >> 7, t = (n >> 5) & 3, o = n & 31;
  const float* bp =
      pre_w + ((size_t)(l * TT + t) * 96 + half * FF) * FF + o;
  v16h B;
#pragma unroll
  for (int i = 0; i < 16; ++i) B[i] = (_Float16)bp[(kbB + i + (i >= 8 ? 8 : 0)) * FF];
  v8f C = {};
  C = wmma16(A, B, C);
  int rr = (lane < 16) ? 0 : 8;
#pragma unroll
  for (int i = 0; i < 8; ++i) P[(size_t)(m0 + i + rr) * 256 + n] = C[i];
}

// -------------------------------------------------- aggregation buffers
__global__ void k_init_agg(float* sum, float* sq, int* mn, int* mx) {
  int i = blockIdx.x * blockDim.x + threadIdx.x;
  if (i >= NN * 128) return;
  sum[i] = 0.f;
  sq[i]  = 0.f;
  mn[i]  = f2ord(3.402823466e38f);
  mx[i]  = f2ord(-3.402823466e38f);
}

// ------------- edge scatter: m = P[dst,:128] + P[src,128:] + cL[attr]
// one wave per edge: indices are wave-uniform -> scalar loads, and the two
// P gathers become single contiguous 512B wave transactions.
__global__ void k_edges(const int* __restrict__ src, const int* __restrict__ dst,
                        const int* __restrict__ attr, const float* __restrict__ P,
                        const float* __restrict__ cL, float* sum, float* sq,
                        int* mn, int* mx) {
  int e = __builtin_amdgcn_readfirstlane(blockIdx.x * 8 + (threadIdx.x >> 5));
  int lane = threadIdx.x & 31;
  int s = __builtin_amdgcn_readfirstlane(src[e]);
  int d = __builtin_amdgcn_readfirstlane(dst[e]);
  int a = __builtin_amdgcn_readfirstlane(attr[e]);
  const float4 p1 = *(const float4*)(P + (size_t)d * 256 + lane * 4);
  const float4 p2 = *(const float4*)(P + (size_t)s * 256 + 128 + lane * 4);
  const float4 cc = *(const float4*)(cL + a * 128 + lane * 4);
  float m[4] = {p1.x + p2.x + cc.x, p1.y + p2.y + cc.y, p1.z + p2.z + cc.z,
                p1.w + p2.w + cc.w};
  size_t base = (size_t)d * 128 + lane * 4;
#pragma unroll
  for (int j = 0; j < 4; ++j) {
    atomicAdd(&sum[base + j], m[j]);
    atomicAdd(&sq[base + j], m[j] * m[j]);
    atomicMax(&mx[base + j], f2ord(m[j]));
    atomicMin(&mn[base + j], f2ord(m[j]));
  }
}

// ---- post MLP: o1 = h@WX + D1 + amp*D2 + att*D3 + post_b   (WMMA)
// post_w for the layer (53KB) is staged once per block into LDS.
__global__ void k_gemm_post(const float* h, const float* sum, const float* sq,
                            const int* mn, const int* mx, const float* dinv,
                            const float* hasf, const float* amp,
                            const float* att, const float* post_w,
                            const float* post_b, float* o1, int l) {
  __shared__ float sPW[TT * 416 * 8];  // 13312 floats = 53KB
  for (int i = threadIdx.x; i < TT * 416 * 8; i += 256)
    sPW[i] = post_w[(size_t)l * TT * 416 * 8 + i];
  __syncthreads();

  int wid = (blockIdx.x * blockDim.x + threadIdx.x) >> 5;
  if (wid >= 3125 * 2) return;  // wave-uniform exit (no further barriers)
  int lane = threadIdx.x & 31;
  int mt = wid >> 1, nt = wid & 1;
  int m0 = mt * 16;
  int r = lane & 15;
  int kbA = (lane < 16) ? 0 : 8;
  int kbB = (lane < 16) ? 0 : 8;
  int rr = (lane < 16) ? 0 : 8;
  int n = nt * 16 + r;            // output col 0..31
  int tn = n >> 3, o = n & 7;     // tower / unit within tower
  const float* pw = sPW + tn * 416 * 8;

  // acc0: h @ WX (dense, K=32)
  v16h A = load_a_row(h + (size_t)(m0 + r) * FF, kbA);
  v16h B;
#pragma unroll
  for (int i = 0; i < 16; ++i) B[i] = (_Float16)pw[(kbB + i + (i >= 8 ? 8 : 0)) * 8 + o];
  v8f c0 = {}, c1 = {}, c2 = {}, c3 = {};
  c0 = wmma16(A, B, c0);

  int row = m0 + r;
  float di = dinv[row], hf = hasf[row];
  const float* srow = sum + (size_t)row * 128;
  const float* qrow = sq + (size_t)row * 128;
  const int* mnrow = mn + (size_t)row * 128;
  const int* mxrow = mx + (size_t)row * 128;

  // block-diagonal towers: only K-chunks for towers {2*nt, 2*nt+1} matter
  for (int kk = nt * 8; kk < nt * 8 + 8; ++kk) {
    int tk = kk >> 2, st = kk & 3;  // tower of this K chunk, stat id
    v16h Ak;
#pragma unroll
    for (int i = 0; i < 16; ++i) {
      int f = (i < 8) ? (kbA + i) : (kbA + 16 + i - 8);
      int col = tk * 32 + f;
      float v;
      if (st == 0)
        v = srow[col] * di;
      else if (st == 1)
        v = (hf != 0.f) ? ord2f(mnrow[col]) : 0.f;
      else if (st == 2)
        v = (hf != 0.f) ? ord2f(mxrow[col]) : 0.f;
      else {
        float me = srow[col] * di;
        float q  = qrow[col] * di;
        v = sqrtf(fmaxf(q - me * me, 0.f) + EPSF);
      }
      Ak[i] = (_Float16)v;
    }
    bool on = (tk == tn);  // per-lane block-diagonal mask
    int rb1 = 32 + st * 32, rb2 = 160 + st * 32, rb3 = 288 + st * 32;
    v16h B1, B2, B3;
#pragma unroll
    for (int i = 0; i < 16; ++i) {
      int f = kbB + i + (i >= 8 ? 8 : 0);
      B1[i] = (_Float16)(on ? pw[(rb1 + f) * 8 + o] : 0.f);
      B2[i] = (_Float16)(on ? pw[(rb2 + f) * 8 + o] : 0.f);
      B3[i] = (_Float16)(on ? pw[(rb3 + f) * 8 + o] : 0.f);
    }
    c1 = wmma16(Ak, B1, c1);
    c2 = wmma16(Ak, B2, c2);
    c3 = wmma16(Ak, B3, c3);
  }
  float bb = post_b[(size_t)(l * TT + tn) * 8 + o];
#pragma unroll
  for (int i = 0; i < 8; ++i) {
    int orow = m0 + i + rr;
    o1[(size_t)orow * 32 + n] =
        c0[i] + c1[i] + amp[orow] * c2[i] + att[orow] * c3[i] + bb;
  }
}

__global__ void k_zero_bn(float* bn) {
  if (threadIdx.x < 64) bn[threadIdx.x] = 0.f;
}

// ---- o = o1 @ lin_w + lin_b  (WMMA) + BN statistics via LDS reduce.
// lin_w layer slice (32x32 f32) is DMA'd into LDS by the Tensor Data Mover.
__global__ void k_lin(const float* o1, const float* lin_w, const float* lin_b,
                      float* out, float* bn, int l) {
  __shared__ float sW[32 * 32];
  __shared__ float s_sum[32], s_sq[32];
  if (threadIdx.x < 32) {
    s_sum[threadIdx.x] = 0.f;
    s_sq[threadIdx.x]  = 0.f;
  }
  if (threadIdx.x < 32) {  // wave 0 issues one TDM descriptor for the block
    unsigned long long ga = (unsigned long long)(const void*)(lin_w + l * 1024);
    unsigned lds_off = (unsigned)(unsigned long long)(const void*)&sW[0];
    // D# group0: count=1, lds_addr, global_addr[56:0], type=2
    v4u g0 = {1u, lds_off, (unsigned)ga,
              (unsigned)((ga >> 32) & 0x01ffffffu) | (2u << 30)};
    // D# group1: data_size=4B; tensor 32x32; tile 32x32; dim0 stride 32
    v8i g1 = {(int)(2u << 16), (int)(32u << 16), (int)(32u << 16),
              (int)(32u << 16), 32, 32, 0, 0};
    v4i g2 = {0, 0, 0, 0};
    v4i g3 = {0, 0, 0, 0};
    v8i g4 = {0, 0, 0, 0, 0, 0, 0, 0};
    __builtin_amdgcn_tensor_load_to_lds(g0, g1, g2, g3, g4, 0);
    __builtin_amdgcn_s_wait_tensorcnt(0);
  }
  __syncthreads();

  int wid  = (blockIdx.x * blockDim.x + threadIdx.x) >> 5;
  int lane = threadIdx.x & 31;
  if (wid < 3125 * 2) {
    int mt = wid >> 1, nt = wid & 1;
    int m0 = mt * 16;
    int r = lane & 15;
    int kbA = (lane < 16) ? 0 : 8;
    int kbB = (lane < 16) ? 0 : 8;
    int rr  = (lane < 16) ? 0 : 8;
    int n = nt * 16 + r;
    v16h A = load_a_row(o1 + (size_t)(m0 + r) * 32, kbA);
    v16h B;
#pragma unroll
    for (int i = 0; i < 16; ++i) B[i] = (_Float16)sW[(kbB + i + (i >= 8 ? 8 : 0)) * 32 + n];
    v8f C = {};
    C = wmma16(A, B, C);
    float lb = lin_b[l * 32 + n];
    float ps = 0.f, pq = 0.f;
#pragma unroll
    for (int i = 0; i < 8; ++i) {
      float v = C[i] + lb;
      out[(size_t)(m0 + i + rr) * 32 + n] = v;
      ps += v;
      pq += v * v;
    }
    atomicAdd(&s_sum[n], ps);   // ds_add_f32
    atomicAdd(&s_sq[n], pq);
  }
  __syncthreads();
  if (threadIdx.x < 32) {
    atomicAdd(&bn[threadIdx.x], s_sum[threadIdx.x]);
    atomicAdd(&bn[32 + threadIdx.x], s_sq[threadIdx.x]);
  }
}

__global__ void k_bnrelu(const float* o, const float* bn, const float* bn_g,
                         const float* bn_b, float* h, int l) {
  int i = blockIdx.x * blockDim.x + threadIdx.x;
  if (i >= NN * 32) return;
  int c = i & 31;
  float mu  = bn[c] * (1.f / NN);
  float var = bn[32 + c] * (1.f / NN) - mu * mu;
  float v = (o[i] - mu) * rsqrtf(var + EPSF) * bn_g[l * 32 + c] + bn_b[l * 32 + c];
  h[i] = fmaxf(v, 0.f);
}

// ------------------------------------------------------------- pooling
__global__ void k_zero_pool(float* pool) {
  int i = blockIdx.x * blockDim.x + threadIdx.x;
  if (i < GG * 32) pool[i] = 0.f;
}

__global__ void k_pool(const int* batch, const float* h, float* pool) {
  int i = blockIdx.x * blockDim.x + threadIdx.x;
  if (i >= NN * 32) return;
  int n = i >> 5, c = i & 31;
  atomicAdd(&pool[batch[n] * 32 + c], h[i]);
}

// -------------------------------------------------------------- head MLP
__global__ void k_head(const float* pool, const float* w1, const float* b1,
                       const float* w2, const float* b2, const float* w3,
                       const float* b3, float* out) {
  __shared__ float sw1[32 * 50], sb1[50], sw2[50 * 25], sb2[25], sw3[25], sb3[1];
  for (int i = threadIdx.x; i < 1600; i += 256) sw1[i] = w1[i];
  for (int i = threadIdx.x; i < 1250; i += 256) sw2[i] = w2[i];
  if (threadIdx.x < 50) sb1[threadIdx.x] = b1[threadIdx.x];
  if (threadIdx.x < 25) {
    sb2[threadIdx.x] = b2[threadIdx.x];
    sw3[threadIdx.x] = w3[threadIdx.x];
  }
  if (threadIdx.x == 0) sb3[0] = b3[0];
  __syncthreads();
  int g = threadIdx.x;
  float hin[32];
  for (int c = 0; c < 32; ++c) hin[c] = pool[g * 32 + c];
  float h1[50];
  for (int j = 0; j < 50; ++j) {
    float s = sb1[j];
    for (int c = 0; c < 32; ++c) s += hin[c] * sw1[c * 50 + j];
    h1[j] = fmaxf(s, 0.f);
  }
  float h2[25];
  for (int j = 0; j < 25; ++j) {
    float s = sb2[j];
    for (int c = 0; c < 50; ++c) s += h1[c] * sw2[c * 25 + j];
    h2[j] = fmaxf(s, 0.f);
  }
  float s = sb3[0];
  for (int c = 0; c < 25; ++c) s += h2[c] * sw3[c];
  out[g] = s;
}

// ================================================================= launch
extern "C" void kernel_launch(void* const* d_in, const int* in_sizes, int n_in,
                              void* d_out, int out_size, void* d_ws,
                              size_t ws_size, hipStream_t stream) {
  (void)in_sizes; (void)n_in; (void)out_size; (void)ws_size;
  const int*   x        = (const int*)d_in[0];
  const int*   ei       = (const int*)d_in[1];
  const int*   src      = ei;            // edge_index[0]
  const int*   dst      = ei + EE;       // edge_index[1]
  const int*   attr     = (const int*)d_in[2];
  const int*   batch    = (const int*)d_in[3];
  const float* node_emb = (const float*)d_in[4];
  const float* edge_emb = (const float*)d_in[5];
  const float* enc_w    = (const float*)d_in[6];
  const float* enc_b    = (const float*)d_in[7];
  const float* pre_w    = (const float*)d_in[8];
  const float* pre_b    = (const float*)d_in[9];
  const float* post_w   = (const float*)d_in[10];
  const float* post_b   = (const float*)d_in[11];
  const float* lin_w    = (const float*)d_in[12];
  const float* lin_b    = (const float*)d_in[13];
  const float* bn_g     = (const float*)d_in[14];
  const float* bn_b     = (const float*)d_in[15];
  const float* w1       = (const float*)d_in[16];
  const float* b1       = (const float*)d_in[17];
  const float* w2       = (const float*)d_in[18];
  const float* b2       = (const float*)d_in[19];
  const float* w3       = (const float*)d_in[20];
  const float* b3       = (const float*)d_in[21];
  const float* avg      = (const float*)d_in[22];
  float* out = (float*)d_out;

  char* ws = (char*)d_ws;
  size_t off = 0;
  auto alloc = [&](size_t bytes) -> void* {
    void* p = ws + off;
    off = (off + bytes + 255) & ~(size_t)255;
    return p;
  };
  float* h    = (float*)alloc((size_t)NN * 32 * 4);
  float* P    = (float*)alloc((size_t)NN * 256 * 4);
  float* sum  = (float*)alloc((size_t)NN * 128 * 4);
  float* sq   = (float*)alloc((size_t)NN * 128 * 4);
  int*   mn   = (int*)alloc((size_t)NN * 128 * 4);
  int*   mx   = (int*)alloc((size_t)NN * 128 * 4);
  float* o1   = (float*)alloc((size_t)NN * 32 * 4);
  float* ob   = (float*)alloc((size_t)NN * 32 * 4);
  int*   deg  = (int*)alloc((size_t)NN * 4);
  float* dinv = (float*)alloc((size_t)NN * 4);
  float* amp  = (float*)alloc((size_t)NN * 4);
  float* att  = (float*)alloc((size_t)NN * 4);
  float* hasf = (float*)alloc((size_t)NN * 4);
  float* cL   = (float*)alloc(4 * 128 * 4);
  float* bn   = (float*)alloc(64 * 4);
  float* pool = (float*)alloc((size_t)GG * 32 * 4);

  k_zero_deg<<<(NN + 255) / 256, 256, 0, stream>>>(deg);
  k_deg<<<(EE + 255) / 256, 256, 0, stream>>>(dst, deg);
  k_nodeprep<<<(NN + 255) / 256, 256, 0, stream>>>(deg, avg, dinv, amp, att, hasf);
  k_embed<<<(NN * 32 + 255) / 256, 256, 0, stream>>>(x, node_emb, h);

  for (int l = 0; l < LL; ++l) {
    k_ctab<<<4, 128, 0, stream>>>(edge_emb, enc_w, enc_b, pre_w, pre_b, cL, l);
    k_gemm_pre<<<(3125 * 16) / 8, 256, 0, stream>>>(h, pre_w, P, l);
    k_init_agg<<<(NN * 128) / 256, 256, 0, stream>>>(sum, sq, mn, mx);
    k_edges<<<EE / 8, 256, 0, stream>>>(src, dst, attr, P, cL, sum, sq, mn, mx);
    k_gemm_post<<<(3125 * 2 + 7) / 8, 256, 0, stream>>>(
        h, sum, sq, mn, mx, dinv, hasf, amp, att, post_w, post_b, o1, l);
    k_zero_bn<<<1, 64, 0, stream>>>(bn);
    k_lin<<<(3125 * 2 + 7) / 8, 256, 0, stream>>>(o1, lin_w, lin_b, ob, bn, l);
    k_bnrelu<<<(NN * 32 + 255) / 256, 256, 0, stream>>>(ob, bn, bn_g, bn_b, h, l);
  }

  k_zero_pool<<<(GG * 32 + 255) / 256, 256, 0, stream>>>(pool);
  k_pool<<<(NN * 32 + 255) / 256, 256, 0, stream>>>(batch, h, pool);
  k_head<<<1, 256, 0, stream>>>(pool, w1, b1, w2, b2, w3, b3, out);
}
